// BasicBlock_26594437497370
// MI455X (gfx1250) — matrix operand order repaired and checked by reading, not compiled, not run on hardware
//
#include <hip/hip_runtime.h>

typedef __attribute__((ext_vector_type(16))) _Float16 v16h;
typedef __attribute__((ext_vector_type(8)))  float    v8f;
typedef int v4i_ __attribute__((vector_size(16)));     // b128 payload type

#define N_IMG 32
#define CH    256
#define HW    28
#define HWW   (HW*HW)        // 784
#define NHW   (N_IMG*HWW)    // 25088
#define KDIM  (CH*9)         // 2304, ordered as [r][cin] (r = ky*3+kx)
#define BM    128
#define BN    128
#define BK    32
#define BKP   40             // padded LDS K-stride (80B rows: 16B aligned)
#define NTILES (NHW/BN)      // 196
#define MTILES (CH/BM)       // 2
#define NEL   (CH*NHW)       // 6,422,528

// CDNA5 async global->LDS staging (ASYNCcnt path), guarded so either toolchain compiles.
#if __has_builtin(__builtin_amdgcn_global_load_async_to_lds_b128) && \
    __has_builtin(__builtin_amdgcn_s_wait_asynccnt)
#define USE_ASYNC_LDS 1
#else
#define USE_ASYNC_LDS 0
#endif

// ---------------------------------------------------------------------------
// K0: fold BN params; convert + PERMUTE weights fp32 OIHW -> f16 [m][r][cin];
//     zero the atomic-max scalars (re-run every call -> deterministic)
// ---------------------------------------------------------------------------
__global__ void init_kernel(const float* __restrict__ w1, const float* __restrict__ w2,
                            const float* __restrict__ g1, const float* __restrict__ b1,
                            const float* __restrict__ m1, const float* __restrict__ v1,
                            const float* __restrict__ g2, const float* __restrict__ b2,
                            const float* __restrict__ m2, const float* __restrict__ v2,
                            _Float16* __restrict__ w1h, _Float16* __restrict__ w2h,
                            float* __restrict__ bn1i, float* __restrict__ bn1s,
                            float* __restrict__ bn2i, float* __restrict__ bn2s,
                            float* __restrict__ S) {
    int idx = blockIdx.x * 256 + threadIdx.x;
    if (idx < CH * KDIM) {
        int m   = idx / KDIM;
        int t   = idx - m * KDIM;      // cin*9 + r  (OIHW flat)
        int cin = t / 9;
        int r   = t - cin * 9;
        int dst = m * KDIM + r * CH + cin;   // [m][r][cin]
        w1h[dst] = (_Float16)w1[idx];
        w2h[dst] = (_Float16)w2[idx];
    }
    if (idx < CH) {
        float i1 = g1[idx] * rsqrtf(v1[idx] + 1e-5f);
        bn1i[idx] = i1;
        bn1s[idx] = b1[idx] - m1[idx] * i1;
        float i2 = g2[idx] * rsqrtf(v2[idx] + 1e-5f);
        bn2i[idx] = i2;
        bn2s[idx] = b2[idx] - m2[idx] * i2;
    }
    if (idx < 2) S[idx] = 0.0f;
}

// ---------------------------------------------------------------------------
// Implicit-GEMM conv3x3 (stride 1, pad 1) via WMMA f16 -> f32.
//   M = Cout(256), N = N*H*W(25088), K = 9*Cin(2304) in [r][cin] order.
// Block tile 128x128, 8 waves -> each wave 64x32 (4x2 WMMA frags).
// Double-buffered LDS; async-DMA weight staging; one barrier per K-step.
// ---------------------------------------------------------------------------
template <bool FIRST>
__global__ __launch_bounds__(256)
void conv_wmma_kernel(const float* __restrict__ xf32,
                      const _Float16* __restrict__ actin,
                      const _Float16* __restrict__ wh,
                      const float* __restrict__ bninv,
                      const float* __restrict__ bnsh,
                      const float* __restrict__ resid,
                      _Float16* __restrict__ actout,
                      float* __restrict__ fout,
                      float* __restrict__ Smax) {
    __shared__ __align__(16) _Float16 sA[2][BM * BKP];  // weights  [m][k]
    __shared__ __align__(16) _Float16 sB[2][BN * BKP];  // im2col, transposed [n][k]

    const int tid   = threadIdx.x;
    const int lane  = tid & 31;
    const int wave  = tid >> 5;
    const int tileN = blockIdx.x % NTILES;
    const int tileM = blockIdx.x / NTILES;
    const int mBase = tileM * BM;
    const int nBase = tileN * BN;
    const int waveM = wave & 1;
    const int waveN = wave >> 1;
    const int half  = lane >> 4;
    const int l15   = lane & 15;

    // B-gather invariants: each thread owns ONE output column nn and a
    // CONTIGUOUS K block [16*kk0, 16*kk0+16)  -> two ds_store_b128.
    const int nn    = tid & 127;
    const int kk0   = tid >> 7;            // 0 or 1
    const int n0    = nBase + nn;
    const int batch = n0 / HWW;
    const int rem   = n0 - batch * HWW;
    const int oy    = rem / HW;
    const int ox    = rem - oy * HW;

    auto stage = [&](int kb, _Float16* __restrict__ dA, _Float16* __restrict__ dB) {
        // ---- A tile: 128x32 f16 weights, b128 chunks (async DMA if available)
#pragma unroll
        for (int i = 0; i < 2; ++i) {
            int c  = tid + 256 * i;       // 0..511
            int m  = c >> 2;
            int kq = (c & 3) * 8;
            const _Float16* gp = wh + (mBase + m) * KDIM + kb + kq;
            _Float16*       lp = &dA[m * BKP + kq];
#if USE_ASYNC_LDS
            __builtin_amdgcn_global_load_async_to_lds_b128(
                (__attribute__((address_space(1))) v4i_*)gp,
                (__attribute__((address_space(3))) v4i_*)lp,
                0, 0);
#else
            *(uint4*)lp = *(const uint4*)gp;
            if (kb + BK < KDIM)
                __builtin_prefetch(gp + BK, 0, 1);
#endif
        }
        // ---- B tile: im2col gather; r is UNIFORM within this K-step ----
        int r       = kb >> 8;             // ky*3+kx   (SALU)
        int cinBase = (kb & (CH - 1)) + 16 * kk0;
        int ky = r / 3, kx = r - ky * 3;   // uniform -> scalar ops
        int iy = oy + ky - 1;
        int ix = ox + kx - 1;
        bool inb = ((unsigned)iy < (unsigned)HW) & ((unsigned)ix < (unsigned)HW);
        union { uint4 q[2]; _Float16 h[16]; } pk;
        if constexpr (FIRST) {
            int xbase = (batch * CH + cinBase) * HWW + iy * HW + ix;
#pragma unroll
            for (int i = 0; i < 16; ++i) {  // 16 loads, one base + imm offsets
                _Float16 v = (_Float16)0.0f;
                if (inb) v = (_Float16)xf32[xbase + i * HWW];
                pk.h[i] = v;
            }
        } else {
            int abase = cinBase * NHW + batch * HWW + iy * HW + ix;
#pragma unroll
            for (int i = 0; i < 16; ++i) {
                _Float16 v = (_Float16)0.0f;
                if (inb) v = actin[abase + i * NHW];
                pk.h[i] = v;
            }
        }
        _Float16* lb = &dB[nn * BKP + 16 * kk0];
        *(uint4*)(lb)     = pk.q[0];
        *(uint4*)(lb + 8) = pk.q[1];
    };

    v8f acc[4][2];
#pragma unroll
    for (int mi = 0; mi < 4; ++mi)
#pragma unroll
        for (int ni = 0; ni < 2; ++ni)
            acc[mi][ni] = (v8f){0.f, 0.f, 0.f, 0.f, 0.f, 0.f, 0.f, 0.f};

    stage(0, sA[0], sB[0]);                 // prologue -> buffer 0

    for (int kb = 0; kb < KDIM; kb += BK) {
#if USE_ASYNC_LDS
        __builtin_amdgcn_s_wait_asynccnt(0);   // prior stage's DMA done
#endif
        __syncthreads();
        const int cur = (kb >> 5) & 1;
        if (kb + BK < KDIM)                 // stage next while computing current
            stage(kb + BK, sA[cur ^ 1], sB[cur ^ 1]);

        // ---- fragments from LDS (contiguous k = 16h+s in both A and B, two b128 each) ----
        v16h afrag[4], bfrag[2];
#pragma unroll
        for (int mi = 0; mi < 4; ++mi) {
            int m = waveM * 64 + mi * 16 + l15;
            const _Float16* p = &sA[cur][m * BKP + half * 16];
            union { v16h h; uint4 q[2]; } u;
            u.q[0] = *(const uint4*)(p);        // K 16h+0..7
            u.q[1] = *(const uint4*)(p + 8);    // K 16h+8..15
            afrag[mi] = u.h;
        }
#pragma unroll
        for (int ni = 0; ni < 2; ++ni) {
            int n = waveN * 32 + ni * 16 + l15;
            const _Float16* p = &sB[cur][n * BKP + half * 16];
            union { v16h h; uint4 q[2]; } u;
            u.q[0] = *(const uint4*)(p);        // K 16h+0..7
            u.q[1] = *(const uint4*)(p + 8);    // K 16h+8..15
            bfrag[ni] = u.h;
        }
#pragma unroll
        for (int mi = 0; mi < 4; ++mi)
#pragma unroll
            for (int ni = 0; ni < 2; ++ni)
                acc[mi][ni] = __builtin_amdgcn_wmma_f32_16x16x32_f16(
                    false, afrag[mi], false, bfrag[ni],
                    (short)0, acc[mi][ni], false, false);
    }

    // ---- epilogue: BN + (residual) + ReLU6 + global max for STE scale ----
    float lmax = 0.0f;
#pragma unroll
    for (int mi = 0; mi < 4; ++mi) {
#pragma unroll
        for (int ni = 0; ni < 2; ++ni) {
            int n = nBase + waveN * 32 + ni * 16 + l15;
            int obatch = n / HWW;
            int orem   = n - obatch * HWW;
#pragma unroll
            for (int j = 0; j < 8; ++j) {
                int c = mBase + waveM * 64 + mi * 16 + j + 8 * half;
                float v = acc[mi][ni][j] * bninv[c] + bnsh[c];
                if constexpr (FIRST) {
                    v = fminf(fmaxf(v, 0.0f), 6.0f);
                    lmax = fmaxf(lmax, v);
                    actout[c * NHW + n] = (_Float16)v;
                } else {
                    int oidx = (obatch * CH + c) * HWW + orem;
                    v += resid[oidx];
                    v = fminf(fmaxf(v, 0.0f), 6.0f);
                    lmax = fmaxf(lmax, v);
                    fout[oidx] = v;
                }
            }
        }
    }
#pragma unroll
    for (int off = 16; off > 0; off >>= 1)
        lmax = fmaxf(lmax, __shfl_xor(lmax, off, 32));
    if (lane == 0)
        atomicMax((unsigned int*)Smax, __float_as_uint(lmax));  // v >= 0: bit-monotonic
}

// ---------------------------------------------------------------------------
// STE 4-bit quantization (15 levels over [0,S]), rintf = round-half-even.
// Vectorized x4 (pure bandwidth passes).
// ---------------------------------------------------------------------------
__global__ void quant_h_kernel(_Float16* __restrict__ a, const float* __restrict__ Sp) {
    int idx = blockIdx.x * 256 + threadIdx.x;   // quad index
    float S  = *Sp;
    float rs = (S > 0.0f) ? 15.0f / S : 0.0f;
    float is = (S > 0.0f) ? S / 15.0f : 0.0f;
    union { uint2 u; _Float16 h[4]; } d;
    d.u = *(uint2*)(a + (size_t)idx * 4);
#pragma unroll
    for (int j = 0; j < 4; ++j)
        d.h[j] = (_Float16)(rintf((float)d.h[j] * rs) * is);
    *(uint2*)(a + (size_t)idx * 4) = d.u;
}

__global__ void quant_f_kernel(float* __restrict__ a, const float* __restrict__ Sp) {
    int idx = blockIdx.x * 256 + threadIdx.x;   // quad index
    float S  = *Sp;
    float rs = (S > 0.0f) ? 15.0f / S : 0.0f;
    float is = (S > 0.0f) ? S / 15.0f : 0.0f;
    float4 d = *(float4*)(a + (size_t)idx * 4);
    d.x = rintf(d.x * rs) * is;
    d.y = rintf(d.y * rs) * is;
    d.z = rintf(d.z * rs) * is;
    d.w = rintf(d.w * rs) * is;
    *(float4*)(a + (size_t)idx * 4) = d;
}

// ---------------------------------------------------------------------------
extern "C" void kernel_launch(void* const* d_in, const int* in_sizes, int n_in,
                              void* d_out, int out_size, void* d_ws, size_t ws_size,
                              hipStream_t stream) {
    const float* x  = (const float*)d_in[0];
    const float* w1 = (const float*)d_in[1];
    const float* g1 = (const float*)d_in[2];
    const float* b1 = (const float*)d_in[3];
    const float* m1 = (const float*)d_in[4];
    const float* v1 = (const float*)d_in[5];
    const float* w2 = (const float*)d_in[6];
    const float* g2 = (const float*)d_in[7];
    const float* b2 = (const float*)d_in[8];
    const float* m2 = (const float*)d_in[9];
    const float* v2 = (const float*)d_in[10];

    char* ws = (char*)d_ws;
    float*    S    = (float*)ws;                       // S[0]=S1, S[1]=S2
    float*    bn1i = (float*)(ws + 256);
    float*    bn1s = (float*)(ws + 256 + 1024);
    float*    bn2i = (float*)(ws + 256 + 2048);
    float*    bn2s = (float*)(ws + 256 + 3072);
    _Float16* w1h  = (_Float16*)(ws + 4608);
    _Float16* w2h  = (_Float16*)(ws + 4608 + (size_t)CH * KDIM * 2);
    _Float16* act1 = (_Float16*)(ws + 4608 + (size_t)CH * KDIM * 4);
    float*    out  = (float*)d_out;

    // K0: BN fold + weight f16 convert/permute + zero scales
    init_kernel<<<(CH * KDIM + 255) / 256, 256, 0, stream>>>(
        w1, w2, g1, b1, m1, v1, g2, b2, m2, v2,
        w1h, w2h, bn1i, bn1s, bn2i, bn2s, S);

    // K1: conv1 + BN + ReLU6 -> act1 (f16, [C][NHW]), S1 = max
    conv_wmma_kernel<true><<<MTILES * NTILES, 256, 0, stream>>>(
        x, nullptr, w1h, bn1i, bn1s, nullptr, act1, nullptr, &S[0]);

    // K2: STE quantize act1 in place (x4 vectorized)
    quant_h_kernel<<<NEL / 1024, 256, 0, stream>>>(act1, &S[0]);

    // K3: conv2 + BN + residual + ReLU6 -> d_out (fp32 NCHW), S2 = max
    conv_wmma_kernel<false><<<MTILES * NTILES, 256, 0, stream>>>(
        nullptr, act1, w2h, bn2i, bn2s, x, nullptr, out, &S[1]);

    // K4: STE quantize d_out in place (x4 vectorized)
    quant_f_kernel<<<NEL / 1024, 256, 0, stream>>>(out, &S[1]);
}
